// GIN_CNN_12661563588883
// MI455X (gfx1250) — matrix operand order repaired and checked by reading, not compiled, not run on hardware
//
#include <hip/hip_runtime.h>
#include <hip/hip_bf16.h>

#define NND 50000
#define NE  600000
#define FIN 16
#define HD  128
#define GG  512
#define CC  4

typedef __attribute__((ext_vector_type(16))) __bf16         v16bf;
typedef __attribute__((ext_vector_type(8)))  float          v8f;
typedef __attribute__((ext_vector_type(8)))  unsigned short ushort8;

struct BfPair { ushort8 lo; ushort8 hi; };

static __device__ __forceinline__ unsigned short f2bfbits(float f) {
  unsigned u = __builtin_bit_cast(unsigned, f);
  unsigned r = (u + 0x7FFFu + ((u >> 16) & 1u)) >> 16;   // round-to-nearest-even
  return (unsigned short)r;
}
static __device__ __forceinline__ __bf16 f2bf(float f) {
  unsigned short b = f2bfbits(f);
  return __builtin_bit_cast(__bf16, b);
}

// ---------------------------------------------------------------------------
// t[i] = (1+eps) * h[i]
__global__ void init_scaled(float* __restrict__ t, const float* __restrict__ h,
                            const float* __restrict__ epsArr, int epsIdx, int total) {
  int idx = blockIdx.x * blockDim.x + threadIdx.x;
  if (idx < total) t[idx] = (1.0f + epsArr[epsIdx]) * h[idx];
}

// scatter-add: t[dst[e]][c] += h[src[e]][c]   (F = 1<<fshift features)
__global__ void scatter_add(const float* __restrict__ h, const int* __restrict__ src,
                            const int* __restrict__ dst, float* __restrict__ t,
                            int fshift, int total) {
  int idx = blockIdx.x * blockDim.x + threadIdx.x;
  if (idx >= total) return;
  int e = idx >> fshift;
  int c = idx & ((1 << fshift) - 1);
  int s = src[e], d = dst[e];
  atomicAdd(&t[((long)d << fshift) + c], h[((long)s << fshift) + c]);
}

// pack W[K,128] (f32, row-major) into the exact B-fragment VGPR layout, bf16,
// zero-padded to K multiple of 32.  Wp[((kk*8+t)*32+lane)*16+i] = W[kk*32+g*8+(i&7)+(i>>3)*16][t*16+n]
__global__ void pack_w(const float* __restrict__ W, unsigned short* __restrict__ Wp, int K) {
  int KK = (K + 31) >> 5;
  int total = KK * 8 * 32 * 16;
  int idx = blockIdx.x * blockDim.x + threadIdx.x;
  if (idx >= total) return;
  int i  = idx & 15;
  int L  = (idx >> 4) & 31;
  int t  = (idx >> 9) & 7;
  int kk = idx >> 12;
  int g = L >> 4, n = L & 15;
  int k = kk * 32 + g * 8 + (i & 7) + ((i >> 3) << 4);
  float v = (k < K) ? W[(long)k * HD + t * 16 + n] : 0.0f;
  Wp[idx] = f2bfbits(v);
}

// ---------------------------------------------------------------------------
// One wave computes a 16x128 strip of D = relu(A[16xK] * W[Kx128] + bias).
// A_IS_F32: A source is f32 (converted in-register), else bf16 (ushort).
// OUT_BF16: write bf16, else f32.
// K is compile time (16 or 128) so the K-loop fully unrolls and the layer-0
// zero-padding folds to constants.
template <bool A_IS_F32, bool OUT_BF16, int K>
__global__ __launch_bounds__(128) void gin_gemm(const void* __restrict__ Asrc,
                                                const unsigned short* __restrict__ Wp,
                                                const float* __restrict__ bias,
                                                void* __restrict__ Dst,
                                                int nstrips) {
  int wid   = threadIdx.x >> 5;
  int strip = blockIdx.x * (blockDim.x >> 5) + wid;
  if (strip >= nstrips) return;                  // wave-uniform
  int lane = threadIdx.x & 31;
  int g = lane >> 4, m = lane & 15;
  int row = strip * 16 + m;

  v8f acc[8];
#pragma unroll
  for (int t = 0; t < 8; ++t)
#pragma unroll
    for (int j = 0; j < 8; ++j) acc[t][j] = 0.0f;

  constexpr int KK = (K + 31) >> 5;
#pragma unroll
  for (int kk = 0; kk < KK; ++kk) {
    v16bf a;
    int kbase = kk * 32 + g * 8;
    if constexpr (A_IS_F32) {
      const float* ap = (const float*)Asrc + (long)row * K;
      if (kk + 1 < KK)
        __builtin_prefetch(ap + kbase + 32, 0, 3);   // next K-step A line
#pragma unroll
      for (int i = 0; i < 8; ++i) {
        int k0 = kbase + i, k1 = kbase + 16 + i;
        float v0 = ap[k0 < K ? k0 : 0];
        float v1 = ap[k1 < K ? k1 : 0];
        a[i]     = f2bf(k0 < K ? v0 : 0.0f);
        a[i + 8] = f2bf(k1 < K ? v1 : 0.0f);
      }
    } else {
      const unsigned short* ap = (const unsigned short*)Asrc + (long)row * K + kbase;
      BfPair p;
      p.lo = *(const ushort8*)(ap);
      p.hi = *(const ushort8*)(ap + 16);
      a = __builtin_bit_cast(v16bf, p);
    }
#pragma unroll
    for (int t = 0; t < 8; ++t) {
      v16bf b = *(const v16bf*)(Wp + (((kk * 8 + t) * 32 + lane) << 4));
      acc[t] = __builtin_amdgcn_wmma_f32_16x16x32_bf16(
          false, a, false, b, (short)0, acc[t], false, false);
    }
  }

#pragma unroll
  for (int t = 0; t < 8; ++t) {
    int col = t * 16 + m;
    float bv = bias[col];
#pragma unroll
    for (int j = 0; j < 8; ++j) {
      int r = strip * 16 + g * 8 + j;        // C/D layout: M = 8*g + j
      float v = acc[t][j] + bv;
      v = v > 0.0f ? v : 0.0f;               // ReLU
      if constexpr (OUT_BF16)
        ((unsigned short*)Dst)[(long)r * HD + col] = f2bfbits(v);
      else
        ((float*)Dst)[(long)r * HD + col] = v;
    }
  }
}

// ---------------------------------------------------------------------------
__global__ void bn_zero(float* a, float* b) {
  int i = threadIdx.x;
  if (i < HD) { a[i] = 0.0f; b[i] = 0.0f; }
}

__global__ void bn_stats(const float* __restrict__ Z, float* __restrict__ sums,
                         float* __restrict__ sumq, int Nrows) {
  int c = threadIdx.x;                   // 128 threads, one column each
  int r0 = blockIdx.x * 128;
  int r1 = r0 + 128; if (r1 > Nrows) r1 = Nrows;
  float s = 0.0f, q = 0.0f;
  for (int r = r0; r < r1; ++r) {
    float v = Z[(long)r * HD + c];
    s += v; q += v * v;
  }
  atomicAdd(&sums[c], s);
  atomicAdd(&sumq[c], q);
}

__global__ void bn_coeff(const float* sums, const float* sumq, const float* gamma,
                         const float* beta, float* A, float* B, int Nrows) {
  int c = threadIdx.x;
  float inv_n = 1.0f / (float)Nrows;
  float mu  = sums[c] * inv_n;
  float var = sumq[c] * inv_n - mu * mu;
  float inv = rsqrtf(var + 1e-5f);
  float a = gamma[c] * inv;
  A[c] = a;
  B[c] = beta[c] - a * mu;
}

__global__ void bn_apply(float* __restrict__ h, const float* __restrict__ z,
                         const float* __restrict__ A, const float* __restrict__ B, int total) {
  int idx = blockIdx.x * blockDim.x + threadIdx.x;
  if (idx >= total) return;
  int c = idx & (HD - 1);
  h[idx] = z[idx] * A[c] + B[c];
}

// ---------------------------------------------------------------------------
__global__ void zero_f32(float* p, int n) {
  int i = blockIdx.x * blockDim.x + threadIdx.x;
  if (i < n) p[i] = 0.0f;
}

__global__ void pool_sum(const float* __restrict__ h, const int* __restrict__ batch,
                         float* __restrict__ pooled, float* __restrict__ cnt, int total) {
  int idx = blockIdx.x * blockDim.x + threadIdx.x;
  if (idx >= total) return;
  int n = idx >> 7, c = idx & (HD - 1);
  int b = batch[n];
  atomicAdd(&pooled[(long)b * HD + c], h[idx]);
  if (c == 0) atomicAdd(&cnt[b], 1.0f);
}

// ---------------------------------------------------------------------------
// CNN branch: one block (128 threads) per graph.
__global__ void cnn_kernel(const float* __restrict__ MD,
                           const float* __restrict__ W1, const float* __restrict__ b1,
                           const float* __restrict__ W2, const float* __restrict__ b2,
                           const float* __restrict__ fcW, const float* __restrict__ fcB,
                           float* __restrict__ xmd) {
  __shared__ float in[144];        // 12x12
  __shared__ float c1[6 * 144];    // 6x12x12
  __shared__ float p1[6 * 36];     // 6x6x6
  __shared__ float m2[64];
  int g = blockIdx.x, tid = threadIdx.x;

  for (int i = tid; i < 144; i += 128) in[i] = MD[(long)g * 144 + i];
  __syncthreads();

  // conv1 5x5, pad 2 -> [6,12,12]
  for (int o = tid; o < 6 * 144; o += 128) {
    int ch = o / 144, r = o % 144, y = r / 12, xx = r % 12;
    float s = b1[ch];
    for (int dy = 0; dy < 5; ++dy)
      for (int dx = 0; dx < 5; ++dx) {
        int iy = y + dy - 2, ix = xx + dx - 2;
        if (iy >= 0 && iy < 12 && ix >= 0 && ix < 12)
          s += in[iy * 12 + ix] * W1[ch * 25 + dy * 5 + dx];
      }
    c1[o] = s;
  }
  __syncthreads();

  // maxpool 2x2 + relu -> [6,6,6]
  for (int o = tid; o < 6 * 36; o += 128) {
    int ch = o / 36, r = o % 36, y = r / 6, xx = r % 6;
    const float* base = &c1[ch * 144 + (2 * y) * 12 + 2 * xx];
    float mx = base[0];
    mx = fmaxf(mx, base[1]); mx = fmaxf(mx, base[12]); mx = fmaxf(mx, base[13]);
    p1[o] = fmaxf(mx, 0.0f);
  }
  __syncthreads();

  // conv2 5x5 valid -> [16,2,2], relu
  if (tid < 64) {
    int co = tid / 4, r = tid % 4, y = r / 2, xx = r % 2;
    float s = b2[co];
    for (int ci = 0; ci < 6; ++ci)
      for (int dy = 0; dy < 5; ++dy)
        for (int dx = 0; dx < 5; ++dx)
          s += p1[ci * 36 + (y + dy) * 6 + (xx + dx)] * W2[((co * 6 + ci) * 5 + dy) * 5 + dx];
    m2[tid] = fmaxf(s, 0.0f);
  }
  __syncthreads();

  // fc1 64->64, relu
  if (tid < 64) {
    float s = fcB[tid];
    for (int k = 0; k < 64; ++k) s += m2[k] * fcW[k * 64 + tid];
    xmd[(long)g * 64 + tid] = fmaxf(s, 0.0f);
  }
}

// ---------------------------------------------------------------------------
// readout: one block (128 threads) per graph
__global__ void readout(const float* __restrict__ pooled, const float* __restrict__ cnt,
                        const float* __restrict__ xmd,
                        const float* __restrict__ l1W, const float* __restrict__ l1b,
                        const float* __restrict__ l2W, const float* __restrict__ l2b,
                        const float* __restrict__ l3W, const float* __restrict__ l3b,
                        float* __restrict__ out) {
  __shared__ float p[HD];
  __shared__ float cat[HD + 64];
  __shared__ float o2[64];
  __shared__ float o3[CC];
  int g = blockIdx.x, tid = threadIdx.x;

  float c = fmaxf(cnt[g], 1.0f);
  p[tid] = pooled[(long)g * HD + tid] / c;
  __syncthreads();

  float s = l1b[tid];
  for (int k = 0; k < HD; ++k) s += p[k] * l1W[k * HD + tid];
  cat[tid] = fmaxf(s, 0.0f);
  if (tid < 64) cat[HD + tid] = xmd[(long)g * 64 + tid];
  __syncthreads();

  if (tid < 64) {
    float s2 = l2b[tid];
    for (int k = 0; k < HD + 64; ++k) s2 += cat[k] * l2W[k * 64 + tid];
    o2[tid] = fmaxf(s2, 0.0f);
  }
  __syncthreads();

  if (tid < CC) {
    float s3 = l3b[tid];
    for (int k = 0; k < 64; ++k) s3 += o2[k] * l3W[k * CC + tid];
    o3[tid] = s3;
  }
  __syncthreads();

  if (tid == 0) {
    float mx = o3[0];
    for (int i = 1; i < CC; ++i) mx = fmaxf(mx, o3[i]);
    float se = 0.0f;
    for (int i = 0; i < CC; ++i) se += expf(o3[i] - mx);
    float lse = mx + logf(se);
    for (int i = 0; i < CC; ++i) out[(long)g * CC + i] = o3[i] - lse;
  }
}

// ---------------------------------------------------------------------------
extern "C" void kernel_launch(void* const* d_in, const int* in_sizes, int n_in,
                              void* d_out, int out_size, void* d_ws, size_t ws_size,
                              hipStream_t stream) {
  (void)in_sizes; (void)n_in; (void)out_size; (void)ws_size;
  const float* x     = (const float*)d_in[0];
  const int*   ei    = (const int*)d_in[1];
  const int*   batch = (const int*)d_in[2];
  const float* MD    = (const float*)d_in[3];
  const float* eps0  = (const float*)d_in[4];
  const float* W1_0  = (const float*)d_in[5];
  const float* b1_0  = (const float*)d_in[6];
  const float* W2_0  = (const float*)d_in[7];
  const float* b2_0  = (const float*)d_in[8];
  const float* g_0   = (const float*)d_in[9];
  const float* be_0  = (const float*)d_in[10];
  const float* eps_l = (const float*)d_in[11];
  const float* W1_l  = (const float*)d_in[12];
  const float* b1_l  = (const float*)d_in[13];
  const float* W2_l  = (const float*)d_in[14];
  const float* b2_l  = (const float*)d_in[15];
  const float* g_l   = (const float*)d_in[16];
  const float* be_l  = (const float*)d_in[17];
  const float* l1W   = (const float*)d_in[18];
  const float* l1b   = (const float*)d_in[19];
  const float* l2W   = (const float*)d_in[20];
  const float* l2b   = (const float*)d_in[21];
  const float* l3W   = (const float*)d_in[22];
  const float* l3b   = (const float*)d_in[23];
  const float* c1W   = (const float*)d_in[24];
  const float* c1b   = (const float*)d_in[25];
  const float* c2W   = (const float*)d_in[26];
  const float* c2b   = (const float*)d_in[27];
  const float* fcW   = (const float*)d_in[28];
  const float* fcb   = (const float*)d_in[29];
  const int* srcI = ei;
  const int* dstI = ei + NE;

  // workspace carve (256B aligned slots)
  char* ws = (char*)d_ws;
  size_t off = 0;
  auto take = [&](size_t bytes) -> char* {
    char* p = ws + off;
    off = (off + bytes + 255) & ~(size_t)255;
    return p;
  };
  float*          hbuf   = (float*)take((size_t)NND * HD * 4);   // h activations
  float*          tz     = (float*)take((size_t)NND * HD * 4);   // t (pre-MLP) / Z (post-GEMM2), reused
  unsigned short* ybf    = (unsigned short*)take((size_t)NND * HD * 2);
  unsigned short* wp1    = (unsigned short*)take((size_t)HD * HD * 2);
  unsigned short* wp2    = (unsigned short*)take((size_t)HD * HD * 2);
  float*          sums   = (float*)take(HD * 4);
  float*          sumq   = (float*)take(HD * 4);
  float*          bnA    = (float*)take(HD * 4);
  float*          bnB    = (float*)take(HD * 4);
  float*          pooled = (float*)take((size_t)GG * HD * 4);
  float*          cntb   = (float*)take(GG * 4);
  float*          xmd    = (float*)take((size_t)GG * 64 * 4);

  // CNN branch (independent)
  cnn_kernel<<<GG, 128, 0, stream>>>(MD, c1W, c1b, c2W, c2b, fcW, fcb, xmd);

  const int strips = NND / 16;               // 3125
  const int gemmBlocks = (strips + 3) / 4;   // 4 waves / block

  for (int layer = 0; layer < 5; ++layer) {
    int F      = (layer == 0) ? FIN : HD;
    int fshift = (layer == 0) ? 4 : 7;
    const float* hin  = (layer == 0) ? x : hbuf;
    const float* epsP = (layer == 0) ? eps0 : eps_l;
    int epsIdx = (layer == 0) ? 0 : (layer - 1);
    const float* W1 = (layer == 0) ? W1_0 : (W1_l + (size_t)(layer - 1) * HD * HD);
    const float* B1 = (layer == 0) ? b1_0 : (b1_l + (layer - 1) * HD);
    const float* W2 = (layer == 0) ? W2_0 : (W2_l + (size_t)(layer - 1) * HD * HD);
    const float* B2 = (layer == 0) ? b2_0 : (b2_l + (layer - 1) * HD);
    const float* GA = (layer == 0) ? g_0 : (g_l + (layer - 1) * HD);
    const float* BE = (layer == 0) ? be_0 : (be_l + (layer - 1) * HD);

    int totalNF = NND * F;
    init_scaled<<<(totalNF + 255) / 256, 256, 0, stream>>>(tz, hin, epsP, epsIdx, totalNF);
    int totalEF = NE * F;
    scatter_add<<<(totalEF + 255) / 256, 256, 0, stream>>>(hin, srcI, dstI, tz, fshift, totalEF);

    int KK1 = (F + 31) / 32;
    pack_w<<<(KK1 * 8 * 32 * 16 + 255) / 256, 256, 0, stream>>>(W1, wp1, F);
    pack_w<<<(4 * 8 * 32 * 16 + 255) / 256, 256, 0, stream>>>(W2, wp2, HD);

    // GEMM1: Y(bf16) = relu(t @ W1 + b1)
    if (layer == 0)
      gin_gemm<true, true, FIN><<<gemmBlocks, 128, 0, stream>>>(tz, wp1, B1, ybf, strips);
    else
      gin_gemm<true, true, HD><<<gemmBlocks, 128, 0, stream>>>(tz, wp1, B1, ybf, strips);
    // GEMM2: Z(f32)  = relu(Y @ W2 + b2)   (overwrites tz — GEMM1 already consumed it)
    gin_gemm<false, false, HD><<<gemmBlocks, 128, 0, stream>>>(ybf, wp2, B2, tz, strips);

    // BatchNorm over nodes
    bn_zero<<<1, 128, 0, stream>>>(sums, sumq);
    bn_stats<<<(NND + 127) / 128, 128, 0, stream>>>(tz, sums, sumq, NND);
    bn_coeff<<<1, 128, 0, stream>>>(sums, sumq, GA, BE, bnA, bnB, NND);
    bn_apply<<<(NND * HD + 255) / 256, 256, 0, stream>>>(hbuf, tz, bnA, bnB, NND * HD);
  }

  // global mean pool
  zero_f32<<<(GG * HD + 255) / 256, 256, 0, stream>>>(pooled, GG * HD);
  zero_f32<<<(GG + 255) / 256, 256, 0, stream>>>(cntb, GG);
  pool_sum<<<(NND * HD + 255) / 256, 256, 0, stream>>>(hbuf, batch, pooled, cntb, NND * HD);

  readout<<<GG, 128, 0, stream>>>(pooled, cntb, xmd, l1W, l1b, l2W, l2b, l3W, l3b,
                                  (float*)d_out);
}
